// RNN_1769526526380
// MI455X (gfx1250) — matrix operand order repaired and checked by reading, not compiled, not run on hardware
//
#include <hip/hip_runtime.h>
#include <hip/hip_bf16.h>
#include <stdint.h>
#include <math.h>

// ---------------------------------------------------------------------------
// RNN scan on MI455X (gfx1250):
//   a_t = tanh(Waa@a_{t-1} + Wax@x_t + ba);  y_t = softmax_batch(Wya@a_t + by)
// Persistent 16-WG kernel, 32 hidden rows per WG, weights split into bf16
// hi/lo pairs resident in LDS (~203 KB/WGP).  All GEMMs run on
// v_wmma_f32_16x16x32_bf16 with 3-term split-precision correction
// (Ahi*Bhi + Alo*Bhi + Ahi*Blo) for ~fp32 accuracy at bf16 WMMA rate.
// Waves 0-7 compute the recurrence for step t, waves 8-15 compute y(t-1)
// concurrently.  One device-wide barrier per step (atomic counter +
// s_sleep spin; also emits the CDNA5 cluster split-barrier, a safe NOP
// when the dispatch is not clustered).
// Workspace: counter(256B) + a-state double buffers (256KB) + x hi/lo
// bf16 in [t][n][k] layout (128MB)  => ~134.5 MB required.
// ---------------------------------------------------------------------------

typedef __bf16 bf16_t;
typedef bf16_t v16bf __attribute__((ext_vector_type(16)));
typedef bf16_t v8bf  __attribute__((ext_vector_type(8)));
typedef float  v8f   __attribute__((ext_vector_type(8)));

#define NA      512
#define MB_     64
#define TX      1024
#define NWG     16
#define ROWS    32          // hidden rows per workgroup
#define WSTRIDE 1032        // padded LDS row stride (K=1024) -> bank-conflict free
#define YSTRIDE 520         // padded LDS row stride (K=512)
#define ZSTRIDE 68

// ---- pre-pass: split a0 into bf16 hi/lo, batch-major [n][r]; zero barrier ---
__global__ void rnn_prep_a0(const float* __restrict__ a0,
                            bf16_t* __restrict__ abh, bf16_t* __restrict__ abl,
                            unsigned* __restrict__ cnt) {
  int tid = blockIdx.x * 256 + threadIdx.x;            // 64*512 elements
  if (tid == 0) *cnt = 0u;
  int n = tid >> 9, r = tid & 511;
  float v = a0[(size_t)r * MB_ + n];
  bf16_t h = (bf16_t)v;
  abh[tid] = h;
  abl[tid] = (bf16_t)(v - (float)h);
}

// ---- pre-pass: split x (n_x, m, t) f32 -> [t][n][k] bf16 hi/lo --------------
__global__ void rnn_prep_x(const float* __restrict__ x,
                           bf16_t* __restrict__ xhi, bf16_t* __restrict__ xlo) {
  size_t tid = (size_t)blockIdx.x * 256 + threadIdx.x; // 1024*64*512 elements
  int k = (int)(tid & 511);
  int n = (int)((tid >> 9) & 63);
  int t = (int)(tid >> 15);
  float v = x[((size_t)k * MB_ + n) * TX + t];
  bf16_t h = (bf16_t)v;
  xhi[tid] = h;
  xlo[tid] = (bf16_t)(v - (float)h);
}

// ---- main persistent kernel -------------------------------------------------
__global__ __launch_bounds__(512, 1) void rnn_persistent(
    const float* __restrict__ Waa, const float* __restrict__ Wax,
    const float* __restrict__ Wya, const float* __restrict__ ba,
    const float* __restrict__ by,
    const bf16_t* __restrict__ xhi, const bf16_t* __restrict__ xlo,
    bf16_t* __restrict__ ab0h, bf16_t* __restrict__ ab0l,
    bf16_t* __restrict__ ab1h, bf16_t* __restrict__ ab1l,
    unsigned* __restrict__ cnt,
    float* __restrict__ out_a, float* __restrict__ out_y) {
  __shared__ bf16_t sWh[ROWS * WSTRIDE];   // [Waa|Wax] slice, bf16 hi
  __shared__ bf16_t sWl[ROWS * WSTRIDE];   // bf16 lo (residual)
  __shared__ bf16_t sYh[ROWS * YSTRIDE];   // Wya slice hi
  __shared__ bf16_t sYl[ROWS * YSTRIDE];   // Wya slice lo
  __shared__ float  sZ[ROWS * ZSTRIDE];    // pre-softmax logits staging
  __shared__ float  sBa[ROWS], sBy[ROWS];

  const int tid = threadIdx.x;
  const int wg  = blockIdx.x;

  // One-time: load + split this WG's weight slices into LDS.
  for (int idx = tid; idx < ROWS * 1024; idx += 512) {
    int i = idx >> 10, k = idx & 1023;
    float v = (k < 512) ? Waa[(size_t)(wg * ROWS + i) * 512 + k]
                        : Wax[(size_t)(wg * ROWS + i) * 512 + (k - 512)];
    bf16_t h = (bf16_t)v;
    sWh[i * WSTRIDE + k] = h;
    sWl[i * WSTRIDE + k] = (bf16_t)(v - (float)h);
  }
  for (int idx = tid; idx < ROWS * 512; idx += 512) {
    int i = idx >> 9, k = idx & 511;
    float v = Wya[(size_t)(wg * ROWS + i) * 512 + k];
    bf16_t h = (bf16_t)v;
    sYh[i * YSTRIDE + k] = h;
    sYl[i * YSTRIDE + k] = (bf16_t)(v - (float)h);
  }
  if (tid < ROWS) { sBa[tid] = ba[wg * ROWS + tid]; sBy[tid] = by[wg * ROWS + tid]; }
  __syncthreads();

  const int wave   = tid >> 5, lane = tid & 31;
  const int laneHi = lane >> 4, l15 = lane & 15;
  const bool isA   = (wave < 8);                 // waves 0-7: recurrence; 8-15: y
  const int ws2    = isA ? wave : (wave - 8);
  const int mi     = ws2 >> 2, ni = ws2 & 3;     // 2x4 grid of 16x16 tiles
  const int n      = ni * 16 + l15;              // output column (batch index)
  const int arow   = mi * 16 + l15;              // A-operand row (local)
  const int rb     = mi * 16 + laneHi * 8;       // first D row (local) this lane

  // A operand (16x32 bf16, M x K): lane holds 16 contiguous K at
  // K = k0 + laneHi*16 (matches B operand order).
  auto ldsA = [&](const bf16_t* base, int stride, int k0) -> v16bf {
    const bf16_t* p = base + arow * stride + k0 + laneHi * 16;
    v8bf a = *(const v8bf*)p;
    v8bf b = *(const v8bf*)(p + 8);
    return __builtin_shufflevector(a, b, 0, 1, 2, 3, 4, 5, 6, 7,
                                   8, 9, 10, 11, 12, 13, 14, 15);
  };
  // B operand (32x16 bf16, K x N): lane holds 16 contiguous K at
  // K = k0 + laneHi*16; source laid out batch-major [n][k].
  auto gB = [&](const bf16_t* p) -> v16bf {
    v8bf a = *(const v8bf*)p;
    v8bf b = *(const v8bf*)(p + 8);
    return __builtin_shufflevector(a, b, 0, 1, 2, 3, 4, 5, 6, 7,
                                   8, 9, 10, 11, 12, 13, 14, 15);
  };

  for (int step = 0; step <= TX; ++step) {
    const bf16_t* aih = (step & 1) ? ab1h : ab0h;  // holds a(step-1)
    const bf16_t* ail = (step & 1) ? ab1l : ab0l;
    bf16_t* aoh = (step & 1) ? ab0h : ab1h;        // receives a(step)
    bf16_t* aol = (step & 1) ? ab0l : ab1l;

    if (isA && step < TX) {
      // a(step) tile = [Waa|Wax] @ [a(step-1); x(step)], split-precision.
      v8f c = {};
#pragma unroll 4
      for (int k0 = 0; k0 < 512; k0 += 32) {
        v16bf Ah = ldsA(sWh, WSTRIDE, k0);
        v16bf Al = ldsA(sWl, WSTRIDE, k0);
        v16bf Bh = gB(aih + n * 512 + k0 + laneHi * 16);
        v16bf Bl = gB(ail + n * 512 + k0 + laneHi * 16);
        c = __builtin_amdgcn_wmma_f32_16x16x32_bf16(false, Ah, false, Bh, (short)0, c, false, false);
        c = __builtin_amdgcn_wmma_f32_16x16x32_bf16(false, Al, false, Bh, (short)0, c, false, false);
        c = __builtin_amdgcn_wmma_f32_16x16x32_bf16(false, Ah, false, Bl, (short)0, c, false, false);
      }
      const size_t xbase = ((size_t)step * MB_ + n) * 512;
      if (step + 1 < TX) {  // global_prefetch_b8 next timestep's x stream
        __builtin_prefetch(xhi + xbase + (size_t)MB_ * 512, 0, 0);
        __builtin_prefetch(xlo + xbase + (size_t)MB_ * 512, 0, 0);
      }
#pragma unroll 4
      for (int k0 = 0; k0 < 512; k0 += 32) {
        v16bf Ah = ldsA(sWh, WSTRIDE, 512 + k0);
        v16bf Al = ldsA(sWl, WSTRIDE, 512 + k0);
        v16bf Bh = gB(xhi + xbase + k0 + laneHi * 16);
        v16bf Bl = gB(xlo + xbase + k0 + laneHi * 16);
        c = __builtin_amdgcn_wmma_f32_16x16x32_bf16(false, Ah, false, Bh, (short)0, c, false, false);
        c = __builtin_amdgcn_wmma_f32_16x16x32_bf16(false, Al, false, Bh, (short)0, c, false, false);
        c = __builtin_amdgcn_wmma_f32_16x16x32_bf16(false, Ah, false, Bl, (short)0, c, false, false);
      }
      // bias + tanh, write f32 output and bf16 hi/lo state for next step.
      float cv[8];
      v8bf h8 = {}, l8 = {};
#pragma unroll
      for (int vg = 0; vg < 8; ++vg) {
        float v = tanhf(c[vg] + sBa[rb + vg]);
        cv[vg] = v;
        bf16_t h = (bf16_t)v;
        h8[vg] = h;
        l8[vg] = (bf16_t)(v - (float)h);
      }
      const int rg = wg * ROWS + rb;
#pragma unroll
      for (int vg = 0; vg < 8; ++vg)
        out_a[((size_t)(rg + vg) * MB_ + n) * TX + step] = cv[vg];
      *(v8bf*)(aoh + n * 512 + rg) = h8;
      *(v8bf*)(aol + n * 512 + rg) = l8;
    }

    if (!isA && step > 0) {
      // logits(step-1) tile = Wya @ a(step-1), split-precision.
      v8f c = {};
#pragma unroll 4
      for (int k0 = 0; k0 < 512; k0 += 32) {
        v16bf Ah = ldsA(sYh, YSTRIDE, k0);
        v16bf Al = ldsA(sYl, YSTRIDE, k0);
        v16bf Bh = gB(aih + n * 512 + k0 + laneHi * 16);
        v16bf Bl = gB(ail + n * 512 + k0 + laneHi * 16);
        c = __builtin_amdgcn_wmma_f32_16x16x32_bf16(false, Ah, false, Bh, (short)0, c, false, false);
        c = __builtin_amdgcn_wmma_f32_16x16x32_bf16(false, Al, false, Bh, (short)0, c, false, false);
        c = __builtin_amdgcn_wmma_f32_16x16x32_bf16(false, Ah, false, Bl, (short)0, c, false, false);
      }
#pragma unroll
      for (int vg = 0; vg < 8; ++vg)
        sZ[(rb + vg) * ZSTRIDE + n] = c[vg] + sBy[rb + vg];
    }
    __syncthreads();

    if (!isA && step > 0) {
      // softmax over the 64-batch axis of each logit row (8 lanes/row).
      const int t2 = tid - 256;
      const int row = t2 >> 3, c0 = (t2 & 7) * 8;
      float z[8], m = -3.402823466e38f;
#pragma unroll
      for (int j = 0; j < 8; ++j) { z[j] = sZ[row * ZSTRIDE + c0 + j]; m = fmaxf(m, z[j]); }
      m = fmaxf(m, __shfl_xor(m, 1, 8));
      m = fmaxf(m, __shfl_xor(m, 2, 8));
      m = fmaxf(m, __shfl_xor(m, 4, 8));
      float s = 0.f;
#pragma unroll
      for (int j = 0; j < 8; ++j) { z[j] = expf(z[j] - m); s += z[j]; }
      s += __shfl_xor(s, 1, 8);
      s += __shfl_xor(s, 2, 8);
      s += __shfl_xor(s, 4, 8);
      const float inv = 1.f / s;
      const size_t rgy = (size_t)(wg * ROWS + row);
#pragma unroll
      for (int j = 0; j < 8; ++j)
        out_y[(rgy * MB_ + c0 + j) * TX + (step - 1)] = z[j] * inv;
    }

    // ---- device-wide step barrier (release -> arrive -> spin -> acquire) ---
    __threadfence();
    __syncthreads();
    if (tid == 0) {
      __hip_atomic_fetch_add(cnt, 1u, __ATOMIC_RELEASE, __HIP_MEMORY_SCOPE_AGENT);
      __builtin_amdgcn_s_cluster_barrier();   // s_barrier_signal/-wait -3 (NOP if unclustered)
      const unsigned tgt = (unsigned)(step + 1) * NWG;
      while (__hip_atomic_load(cnt, __ATOMIC_ACQUIRE, __HIP_MEMORY_SCOPE_AGENT) < tgt)
        __builtin_amdgcn_s_sleep(2);
    }
    __syncthreads();
    __threadfence();
  }
}

extern "C" void kernel_launch(void* const* d_in, const int* in_sizes, int n_in,
                              void* d_out, int out_size, void* d_ws, size_t ws_size,
                              hipStream_t stream) {
  (void)in_sizes; (void)n_in; (void)out_size; (void)ws_size;
  const float* x   = (const float*)d_in[0];
  const float* a0  = (const float*)d_in[1];
  const float* Waa = (const float*)d_in[2];
  const float* Wax = (const float*)d_in[3];
  const float* Wya = (const float*)d_in[4];
  const float* ba  = (const float*)d_in[5];
  const float* by  = (const float*)d_in[6];

  // Workspace layout (~134.5 MB): counter | a double-buffers | x hi | x lo
  char* ws = (char*)d_ws;
  unsigned* cnt = (unsigned*)ws;
  bf16_t* ab0h = (bf16_t*)(ws + 256);
  bf16_t* ab0l = ab0h + (size_t)MB_ * 512;
  bf16_t* ab1h = ab0l + (size_t)MB_ * 512;
  bf16_t* ab1l = ab1h + (size_t)MB_ * 512;
  bf16_t* xhi  = ab1l + (size_t)MB_ * 512;
  bf16_t* xlo  = xhi + (size_t)TX * MB_ * 512;

  float* out_a = (float*)d_out;
  float* out_y = out_a + (size_t)NA * MB_ * TX;

  rnn_prep_a0<<<(MB_ * 512) / 256, 256, 0, stream>>>(a0, ab0h, ab0l, cnt);
  rnn_prep_x<<<((size_t)TX * MB_ * 512) / 256, 256, 0, stream>>>(x, xhi, xlo);
  rnn_persistent<<<NWG, 512, 0, stream>>>(Waa, Wax, Wya, ba, by, xhi, xlo,
                                          ab0h, ab0l, ab1h, ab1l, cnt,
                                          out_a, out_y);
}
